// CausalSelfAttention_15796889715420
// MI455X (gfx1250) — matrix operand order repaired and checked by reading, not compiled, not run on hardware
//
#include <hip/hip_runtime.h>
#include <hip/hip_bf16.h>
#include <stdint.h>

// ---------- types for WMMA ----------
typedef __attribute__((ext_vector_type(16))) __bf16 v16bf;
typedef __attribute__((ext_vector_type(8)))  float  v8f;

union Frag16 { v16bf v; uint4 q[2]; };

__device__ __forceinline__ unsigned short f2bf(float f) {
  union { float f; unsigned u; } x; x.f = f;
  unsigned r = x.u + 0x7FFFu + ((x.u >> 16) & 1u);   // round-to-nearest-even
  return (unsigned short)(r >> 16);
}

__device__ __forceinline__ v16bf ldfrag(const unsigned short* p0, const unsigned short* p1) {
  Frag16 f;
  f.q[0] = *(const uint4*)p0;
  f.q[1] = *(const uint4*)p1;
  return f.v;
}

__device__ __forceinline__ v8f wmma_bf16(v16bf a, v16bf b, v8f c) {
  // D = A(16x32 bf16) * B(32x16 bf16) + C(16x16 f32)
  return __builtin_amdgcn_wmma_f32_16x16x32_bf16(false, a, false, b, (short)0, c, false, false);
}

// ---------- CDNA5 async global->LDS copy (16B per lane), ASYNCcnt tracked ----
__device__ __forceinline__ void async_copy16(const void* gptr, void* ldsptr) {
  unsigned lds = (unsigned)(uintptr_t)ldsptr;  // low 32 bits of generic LDS ptr
  asm volatile("global_load_async_to_lds_b128 %0, %1, off"
               :: "v"(lds), "v"(gptr)
               : "memory");
}

__device__ __forceinline__ void async_wait0() {
  asm volatile("s_wait_asynccnt 0x0" ::: "memory");
}

// ---------- problem constants ----------
#define BB 4
#define TT 2048
#define DM 1024
#define NH 16
#define HD 64
#define MROWS (BB * TT)      // 8192

// ---------- fp32 -> bf16 conversion (8 elems / thread / iter) ----------
__global__ void cvt_f32_bf16(const float* __restrict__ in,
                             unsigned short* __restrict__ out, int n8) {
  for (int i = blockIdx.x * blockDim.x + threadIdx.x; i < n8;
       i += gridDim.x * blockDim.x) {
    const float4* p = (const float4*)in + (size_t)i * 2;
    float4 a = p[0], b = p[1];
    union { unsigned short h[8]; uint4 q; } u;
    u.h[0] = f2bf(a.x); u.h[1] = f2bf(a.y); u.h[2] = f2bf(a.z); u.h[3] = f2bf(a.w);
    u.h[4] = f2bf(b.x); u.h[5] = f2bf(b.y); u.h[6] = f2bf(b.z); u.h[7] = f2bf(b.w);
    *(uint4*)(out + (size_t)i * 8) = u.q;
  }
}

// ---------- NT GEMM: C[M,N] = A[M,K] * B[N,K]^T, A/B bf16, K-major ----------
// block tile 128x128, K-step 32, 256 threads = 8 waves (4 in M x 2 in N),
// double-buffered LDS fed by async global->LDS copies.
template <bool OUT_BF16>
__global__ __launch_bounds__(256) void gemm_nt_bf16(
    const unsigned short* __restrict__ A,
    const unsigned short* __restrict__ B,
    void* __restrict__ Cout, int M, int N, int K) {
  constexpr int LDT = 40;  // padded LDS row stride (elems) -> conflict-free b128
  __shared__ unsigned short As[2][128 * LDT];
  __shared__ unsigned short Bs[2][128 * LDT];

  const int tid  = threadIdx.x;
  const int lane = tid & 31;
  const int wid  = tid >> 5;
  const int wm   = wid >> 1;        // 0..3
  const int wn   = wid & 1;         // 0..1
  const int hw   = lane >> 4;       // half of wave
  const int ln   = lane & 15;

  const int mBase = blockIdx.y * 128;
  const int nBase = blockIdx.x * 128;

  const int ldRow  = tid >> 1;          // 0..127
  const int ldPart = (tid & 1) * 16;    // 0 / 16

  const unsigned short* Aptr = A + (size_t)(mBase + ldRow) * K + ldPart;
  const unsigned short* Bptr = B + (size_t)(nBase + ldRow) * K + ldPart;

  const v8f vzero = {0.f, 0.f, 0.f, 0.f, 0.f, 0.f, 0.f, 0.f};
  v8f acc[2][4];
  for (int i = 0; i < 2; ++i)
    for (int j = 0; j < 4; ++j) acc[i][j] = vzero;

  // issue async loads for k-tile 0 into buffer 0
  {
    async_copy16(Aptr,     &As[0][ldRow * LDT + ldPart]);
    async_copy16(Aptr + 8, &As[0][ldRow * LDT + ldPart + 8]);
    async_copy16(Bptr,     &Bs[0][ldRow * LDT + ldPart]);
    async_copy16(Bptr + 8, &Bs[0][ldRow * LDT + ldPart + 8]);
  }

  const int nk = K / 32;
  for (int kt = 0; kt < nk; ++kt) {
    async_wait0();     // this wave's async loads for tile kt landed in LDS
    __syncthreads();   // all waves' loads landed; prior-buffer reads done
    if (kt + 1 < nk) {
      const int nb = (kt + 1) & 1;
      const unsigned short* Ap = Aptr + (kt + 1) * 32;
      const unsigned short* Bp = Bptr + (kt + 1) * 32;
      async_copy16(Ap,     &As[nb][ldRow * LDT + ldPart]);
      async_copy16(Ap + 8, &As[nb][ldRow * LDT + ldPart + 8]);
      async_copy16(Bp,     &Bs[nb][ldRow * LDT + ldPart]);
      async_copy16(Bp + 8, &Bs[nb][ldRow * LDT + ldPart + 8]);
    }
    const unsigned short* Asb = As[kt & 1];
    const unsigned short* Bsb = Bs[kt & 1];

    // A fragments (16x32): lane m = ln, K split by half-wave
    v16bf aF[2];
#pragma unroll
    for (int i = 0; i < 2; ++i) {
      const unsigned short* p0 = &Asb[(wm * 32 + i * 16 + ln) * LDT + hw * 8];
      aF[i] = ldfrag(p0, p0 + 16);
    }
#pragma unroll
    for (int j = 0; j < 4; ++j) {
      // B fragment (32x16): lane n = ln, k = hw*8 + 0..7 and 16 + hw*8 + 0..7
      const unsigned short* p0 = &Bsb[(wn * 64 + j * 16 + ln) * LDT + hw * 8];
      v16bf bF = ldfrag(p0, p0 + 16);
      acc[0][j] = wmma_bf16(aF[0], bF, acc[0][j]);
      acc[1][j] = wmma_bf16(aF[1], bF, acc[1][j]);
    }
  }

  // epilogue: C layout lane n = ln, VGPR v -> m = v + 8*hw
#pragma unroll
  for (int i = 0; i < 2; ++i)
#pragma unroll
    for (int j = 0; j < 4; ++j)
#pragma unroll
      for (int v = 0; v < 8; ++v) {
        int row = mBase + wm * 32 + i * 16 + hw * 8 + v;
        int col = nBase + wn * 64 + j * 16 + ln;
        if (OUT_BF16)
          ((unsigned short*)Cout)[(size_t)row * N + col] = f2bf(acc[i][j][v]);
        else
          ((float*)Cout)[(size_t)row * N + col] = acc[i][j][v];
      }
}

// ---------- flash attention: one block per (b, h, 128 query rows) ----------
// 8 waves x 16 query rows; key tiles of 64; online softmax; bf16 WMMA.
// Q and K tiles staged via async global->LDS; V transposed through registers.
__global__ __launch_bounds__(256) void attn_kernel(
    const unsigned short* __restrict__ qkv,   // [B*T, 3072] bf16
    unsigned short* __restrict__ outp) {      // [B*T, 1024] bf16
  constexpr int LDK = 72;  // padded LDS stride
  __shared__ unsigned short Qs[128 * LDK];
  __shared__ unsigned short Ks[64 * LDK];
  __shared__ unsigned short Vs[64 * LDK];       // transposed: [d][key]
  __shared__ unsigned short Ps[8][16 * LDK];    // per-wave P tile

  const int blk  = blockIdx.x;
  const int qblk = blk & 15;
  const int h    = (blk >> 4) & 15;
  const int b    = blk >> 8;
  const int q0   = qblk * 128;

  const int tid  = threadIdx.x;
  const int lane = tid & 31;
  const int wid  = tid >> 5;
  const int hw   = lane >> 4;
  const int ln   = lane & 15;

  const unsigned short* Qg = qkv + (size_t)b * TT * 3072 + h * HD;
  const unsigned short* Kg = Qg + 1024;
  const unsigned short* Vg = Qg + 2048;

  // stage the 128x64 Q block (async -> LDS)
  for (int idx = tid; idx < 128 * 8; idx += 256) {
    int r = idx >> 3, s = (idx & 7) * 8;
    async_copy16(Qg + (size_t)(q0 + r) * 3072 + s, &Qs[r * LDK + s]);
  }
  async_wait0();
  __syncthreads();

  // wave-resident Q fragments (d 0..31 and 32..63)
  v16bf qA[2];
#pragma unroll
  for (int g = 0; g < 2; ++g) {
    const unsigned short* p0 = &Qs[(wid * 16 + ln) * LDK + g * 32 + hw * 8];
    qA[g] = ldfrag(p0, p0 + 16);
  }

  const float K1 = 0.125f * 1.44269504088896340736f;  // 1/sqrt(64) * log2(e)
  const v8f vzero = {0.f, 0.f, 0.f, 0.f, 0.f, 0.f, 0.f, 0.f};

  float rowMax[8], rowSum[8];
  v8f o[4];
#pragma unroll
  for (int v = 0; v < 8; ++v) { rowMax[v] = -1e30f; rowSum[v] = 0.f; }
#pragma unroll
  for (int j = 0; j < 4; ++j) o[j] = vzero;

  const int qrow0 = q0 + wid * 16;        // wave's first query row
  const int nkt   = (q0 + 128) >> 6;      // causal key-tile count for block

  for (int kt = 0; kt < nkt; ++kt) {
    const int kbase = kt * 64;
    __syncthreads();  // prior reads of Ks/Vs done
    // stage K tile (async) and transposed V tile (register transpose)
    for (int idx = tid; idx < 64 * 8; idx += 256) {
      int r = idx >> 3, s = (idx & 7) * 8;
      async_copy16(Kg + (size_t)(kbase + r) * 3072 + s, &Ks[r * LDK + s]);
      uint4 vv = *(const uint4*)(Vg + (size_t)(kbase + r) * 3072 + s);
      const unsigned short* pv = (const unsigned short*)&vv;
#pragma unroll
      for (int e = 0; e < 8; ++e) Vs[(s + e) * LDK + r] = pv[e];
    }
    async_wait0();
    __syncthreads();

    if (kbase <= qrow0 + 15) {  // wave-uniform: any unmasked keys for this wave
      float z[4][8];
      float tmax[8];
#pragma unroll
      for (int v = 0; v < 8; ++v) tmax[v] = -INFINITY;

#pragma unroll
      for (int j = 0; j < 4; ++j) {
        v8f s = vzero;
        const unsigned short* kp0 = &Ks[(j * 16 + ln) * LDK + hw * 8];
        s = wmma_bf16(qA[0], ldfrag(kp0, kp0 + 16), s);
        const unsigned short* kp1 = kp0 + 32;
        s = wmma_bf16(qA[1], ldfrag(kp1, kp1 + 16), s);
        const int nglob = kbase + j * 16 + ln;
#pragma unroll
        for (int v = 0; v < 8; ++v) {
          int m = qrow0 + hw * 8 + v;
          float val = (nglob <= m) ? s[v] * K1 : -INFINITY;
          z[j][v] = val;
          tmax[v] = fmaxf(tmax[v], val);
        }
      }
      // row max across the 16 lanes of each half
#pragma unroll
      for (int v = 0; v < 8; ++v) {
        tmax[v] = fmaxf(tmax[v], __shfl_xor(tmax[v], 1, 32));
        tmax[v] = fmaxf(tmax[v], __shfl_xor(tmax[v], 2, 32));
        tmax[v] = fmaxf(tmax[v], __shfl_xor(tmax[v], 4, 32));
        tmax[v] = fmaxf(tmax[v], __shfl_xor(tmax[v], 8, 32));
      }
      float corr[8], lsum[8];
#pragma unroll
      for (int v = 0; v < 8; ++v) {
        float nm = fmaxf(rowMax[v], tmax[v]);  // >= -1e30 floor
        corr[v] = exp2f(rowMax[v] - nm);
        rowMax[v] = nm;
        lsum[v] = 0.f;
      }
#pragma unroll
      for (int j = 0; j < 4; ++j)
#pragma unroll
        for (int v = 0; v < 8; ++v) {
          float p = exp2f(z[j][v] - rowMax[v]);  // -inf -> 0
          z[j][v] = p;
          lsum[v] += p;
        }
#pragma unroll
      for (int v = 0; v < 8; ++v) {
        lsum[v] += __shfl_xor(lsum[v], 1, 32);
        lsum[v] += __shfl_xor(lsum[v], 2, 32);
        lsum[v] += __shfl_xor(lsum[v], 4, 32);
        lsum[v] += __shfl_xor(lsum[v], 8, 32);
        rowSum[v] = rowSum[v] * corr[v] + lsum[v];
      }
#pragma unroll
      for (int j = 0; j < 4; ++j)
#pragma unroll
        for (int v = 0; v < 8; ++v) o[j][v] *= corr[v];

      // write P (bf16) to wave-private LDS, A-fragment-readable layout
      unsigned short* Pw = Ps[wid];
#pragma unroll
      for (int j = 0; j < 4; ++j)
#pragma unroll
        for (int v = 0; v < 8; ++v)
          Pw[(hw * 8 + v) * LDK + j * 16 + ln] = f2bf(z[j][v]);

      // O += P @ V
#pragma unroll
      for (int kc = 0; kc < 2; ++kc) {
        const unsigned short* pp = &Pw[ln * LDK + kc * 32 + hw * 8];
        v16bf aP = ldfrag(pp, pp + 16);
#pragma unroll
        for (int jd = 0; jd < 4; ++jd) {
          const unsigned short* vp =
              &Vs[(jd * 16 + ln) * LDK + kc * 32 + hw * 8];
          o[jd] = wmma_bf16(aP, ldfrag(vp, vp + 16), o[jd]);
        }
      }
    }
  }

  // normalize and store bf16 attn output at [b, q, h*64 + d]
  float inv[8];
#pragma unroll
  for (int v = 0; v < 8; ++v) inv[v] = 1.0f / rowSum[v];
#pragma unroll
  for (int jd = 0; jd < 4; ++jd)
#pragma unroll
    for (int v = 0; v < 8; ++v) {
      int m = qrow0 + hw * 8 + v;
      int d = jd * 16 + ln;
      outp[((size_t)(b * TT + m)) * DM + h * HD + d] = f2bf(o[jd][v] * inv[v]);
    }
}

// ---------- host launch ----------
extern "C" void kernel_launch(void* const* d_in, const int* in_sizes, int n_in,
                              void* d_out, int out_size, void* d_ws,
                              size_t ws_size, hipStream_t stream) {
  (void)in_sizes; (void)n_in; (void)out_size; (void)ws_size;
  const float* x    = (const float*)d_in[0];   // [4,2048,1024]
  const float* wqkv = (const float*)d_in[1];   // [3072,1024]
  const float* wout = (const float*)d_in[2];   // [1024,1024]
  float* out = (float*)d_out;                  // [4,2048,1024]

  const size_t NX  = (size_t)MROWS * DM;       // 8.39M
  const size_t NWQ = (size_t)3 * DM * DM;      // 3.15M
  const size_t NWO = (size_t)DM * DM;          // 1.05M
  const size_t NQK = (size_t)MROWS * 3 * DM;   // 25.2M
  // bf16 workspace layout
  unsigned short* xb    = (unsigned short*)d_ws;
  unsigned short* wqb   = xb + NX;
  unsigned short* wob   = wqb + NWQ;
  unsigned short* qkvb  = wob + NWO;
  unsigned short* attnb = qkvb + NQK;

  cvt_f32_bf16<<<2048, 256, 0, stream>>>(x, xb, (int)(NX / 8));
  cvt_f32_bf16<<<1024, 256, 0, stream>>>(wqkv, wqb, (int)(NWQ / 8));
  cvt_f32_bf16<<<512, 256, 0, stream>>>(wout, wob, (int)(NWO / 8));

  // QKV projection: [8192,1024] x [3072,1024]^T -> bf16 [8192,3072]
  gemm_nt_bf16<true><<<dim3(3 * DM / 128, MROWS / 128), 256, 0, stream>>>(
      xb, wqb, qkvb, MROWS, 3 * DM, DM);

  // causal attention -> bf16 [8192,1024]
  attn_kernel<<<dim3(BB * NH * (TT / 128)), 256, 0, stream>>>(qkvb, attnb);

  // output projection: [8192,1024] x [1024,1024]^T -> fp32 d_out
  gemm_nt_bf16<false><<<dim3(DM / 128, MROWS / 128), 256, 0, stream>>>(
      attnb, wob, out, MROWS, DM, DM);
}
